// HRMLM_34548716929793
// MI455X (gfx1250) — matrix operand order repaired and checked by reading, not compiled, not run on hardware
//
#include <hip/hip_runtime.h>
#include <hip/hip_bf16.h>

#define B_ 64
#define S_ 64
#define V_ 32000
#define E_ 768
#define H_ 1024
#define HH_ 768
#define LOWIN_ (E_ + HH_)   /* 1536 */
#define EPS_ 1e-5f
#define KSTEP_ 64

typedef __attribute__((ext_vector_type(16))) __bf16 bf16x16;
typedef __attribute__((ext_vector_type(8)))  __bf16 bf16x8;
typedef __attribute__((ext_vector_type(8)))  float  f32x8;

// ---------------------------------------------------------------------------
// Weight convert + transpose: W[K][N] f32 (row-major)  ->  Wt[N][K] bf16
// ---------------------------------------------------------------------------
__global__ void k_transpose_bf16(const float* __restrict__ W,
                                 __hip_bfloat16* __restrict__ Wt,
                                 int K, int N) {
  __shared__ float tile[32][33];
  const int tn = blockIdx.x * 32;   // N tile
  const int tk = blockIdx.y * 32;   // K tile
  const int tx = threadIdx.x;       // 0..31
  const int ty = threadIdx.y;       // 0..7
  for (int i = ty; i < 32; i += 8)
    tile[i][tx] = W[(size_t)(tk + i) * N + (tn + tx)];
  __syncthreads();
  for (int i = ty; i < 32; i += 8)
    Wt[(size_t)(tn + i) * K + (tk + tx)] = __float2bfloat16(tile[tx][i]);
}

// ---------------------------------------------------------------------------
// Block reduction helper (blockDim.x == 256)
// ---------------------------------------------------------------------------
__device__ __forceinline__ float blockReduceSum(float v, float* red) {
  const int t = threadIdx.x;
  red[t] = v;
  __syncthreads();
  for (int o = 128; o > 0; o >>= 1) {
    if (t < o) red[t] += red[t + o];
    __syncthreads();
  }
  float r = red[0];
  __syncthreads();
  return r;
}

// ---------------------------------------------------------------------------
// Embedding gather + LayerNorm -> bf16, time-major x[S][B][E]
// ---------------------------------------------------------------------------
__global__ void k_embed_ln(const int* __restrict__ ids,
                           const float* __restrict__ emb,
                           const float* __restrict__ g,
                           const float* __restrict__ bta,
                           __hip_bfloat16* __restrict__ xbf) {
  __shared__ float red[256];
  const int bs = blockIdx.x;
  const int b = bs / S_, s = bs % S_;
  const float* row = emb + (size_t)ids[b * S_ + s] * E_;
  float sum = 0.f, sq = 0.f;
  for (int e = threadIdx.x; e < E_; e += 256) {
    float v = row[e];
    sum += v;
    sq  += v * v;
  }
  sum = blockReduceSum(sum, red);
  sq  = blockReduceSum(sq, red);
  const float m = sum / (float)E_;
  const float rstd = rsqrtf(sq / (float)E_ - m * m + EPS_);
  __hip_bfloat16* o = xbf + ((size_t)s * B_ + b) * E_;
  for (int e = threadIdx.x; e < E_; e += 256)
    o[e] = __float2bfloat16((row[e] - m) * rstd * g[e] + bta[e]);
}

// ---------------------------------------------------------------------------
// Zero-initialize recurrent state (fp32 + bf16 mirrors)
// ---------------------------------------------------------------------------
__global__ void k_zero_state(float* hl, __hip_bfloat16* hlb,
                             float* hh, __hip_bfloat16* hhb) {
  const int i = blockIdx.x * 256 + threadIdx.x;
  if (i < B_ * H_)  { hl[i] = 0.f; hlb[i] = __float2bfloat16(0.f); }
  if (i < B_ * HH_) { hh[i] = 0.f; hhb[i] = __float2bfloat16(0.f); }
}

// ---------------------------------------------------------------------------
// Pack xcat[B][E+HH] = cat(x_t, h_h) in bf16.
// ---------------------------------------------------------------------------
__global__ void k_pack_xcat(const __hip_bfloat16* __restrict__ xs,  // [B][E]
                            const __hip_bfloat16* __restrict__ hh,  // [B][HH]
                            __hip_bfloat16* __restrict__ xc) {      // [B][E+HH]
  const int b = blockIdx.x;
  __hip_bfloat16* o = xc + (size_t)b * LOWIN_;
  for (int i = threadIdx.x; i < E_; i += 256)  o[i]      = xs[(size_t)b * E_ + i];
  for (int i = threadIdx.x; i < HH_; i += 256) o[E_ + i] = hh[(size_t)b * HH_ + i];
}

// ---------------------------------------------------------------------------
// WMMA GEMM:  C[64][N] (fp32, row stride ldc) = A[64][K] bf16 * Bt[N][K] bf16 + bias
//
// grid.x = N/64, block = 128 (4 waves).  Block computes a 64x64 C tile:
//  - wave w owns n-tile w; its B fragment (32B/lane) is reused across all four
//    16-row M tiles -> 4 WMMAs per B fetch.
//  - the shared 64x64 A panel (8KB) is double-buffered in LDS and filled with
//    CDNA5 async copies (global_load_async_to_lds_b128, ASYNCcnt).  The LDS
//    destination operands are derived from the real panel pointers (ptrtoint),
//    which both gives the wave-relative LDS byte address and makes the panels
//    escape, so the "memory"-clobbering asm is known to write them and the
//    plain ds reads below cannot be folded away or hoisted past the barrier.
//  - one s_wait_asynccnt + one barrier per 64-wide K step; next panel's copies
//    are issued right after the barrier and overlap the 8 WMMAs.
// ---------------------------------------------------------------------------
__global__ void k_gemm_bf16_wmma(const __hip_bfloat16* __restrict__ A,
                                 const __hip_bfloat16* __restrict__ Bt,
                                 const float* __restrict__ bias,
                                 float* __restrict__ C,
                                 int K, long ldc) {
  __shared__ __align__(16) __hip_bfloat16 apanel[2][64 * KSTEP_];   // 2 x 8 KB

  const int tid  = threadIdx.x;
  const int lane = tid & 31;
  const int wave = tid >> 5;             // n-tile within block (0..3)
  const int half = lane >> 4;
  const int l15  = lane & 15;
  const long ncol = (long)blockIdx.x * 64 + wave * 16 + l15;

  const __bf16* brow  = reinterpret_cast<const __bf16*>(Bt) + (size_t)ncol * K;
  const __bf16* abase = reinterpret_cast<const __bf16*>(A);

  const unsigned pb[2] = { (unsigned)(size_t)&apanel[0][0],
                           (unsigned)(size_t)&apanel[1][0] };

  // Each thread copies 4x16B chunks per 64x64 panel: chunk c in [0,512):
  // row = c>>3, 8-elem octet = c&7.
  auto stage = [&](int k0, unsigned pbase) {
#pragma unroll
    for (int j = 0; j < 4; ++j) {
      const unsigned c = (unsigned)tid + j * 128u;
      const unsigned row = c >> 3, oct = c & 7;
      const __bf16* g = abase + (size_t)row * K + k0 + oct * 8;
      const unsigned l = pbase + c * 16u;
      asm volatile("global_load_async_to_lds_b128 %0, %1, off"
                   :: "v"(l), "v"(g) : "memory");
    }
  };

  f32x8 acc[4] = {};
  stage(0, pb[0]);

  const int nsteps = K / KSTEP_;
  for (int i = 0; i < nsteps; ++i) {
    const int k0 = i * KSTEP_;
    const int p  = i & 1;

    asm volatile("s_wait_asynccnt 0x0" ::: "memory");  // my copies for panel p done
    __syncthreads();                                   // everyone's copies done;
                                                       // everyone done reading p^1
    if (i + 1 < nsteps) stage(k0 + KSTEP_, pb[p ^ 1]); // overlap with compute

    const __hip_bfloat16* pan = apanel[p];
#pragma unroll
    for (int kc = 0; kc < 2; ++kc) {
      bf16x16 b = *reinterpret_cast<const bf16x16*>(brow + k0 + kc * 32 + half * 16);
      __builtin_prefetch(brow + k0 + 256, 0, 1);       // global_prefetch_b8
#pragma unroll
      for (int mt = 0; mt < 4; ++mt) {
        union { bf16x16 v; bf16x8 h[2]; } a;
        const __hip_bfloat16* arow = pan + (mt * 16 + l15) * KSTEP_ + kc * 32;
        a.h[0] = *reinterpret_cast<const bf16x8*>(arow + half * 16);
        a.h[1] = *reinterpret_cast<const bf16x8*>(arow + half * 16 + 8);
        acc[mt] = __builtin_amdgcn_wmma_f32_16x16x32_bf16(
            false, a.v, false, b, (short)0, acc[mt], false, false);
      }
    }
  }

  const float bv = bias[ncol];
  float* cc = C + ncol;
  const int rbase = half * 8;
#pragma unroll
  for (int mt = 0; mt < 4; ++mt)
#pragma unroll
    for (int r = 0; r < 8; ++r)
      cc[(size_t)(mt * 16 + rbase + r) * ldc] = acc[mt][r] + bv;
}

// ---------------------------------------------------------------------------
// Fused LN(i) + LN(hh) + GRU-style gates + state update (+ bf16 mirror).
// ---------------------------------------------------------------------------
__global__ void k_cell_update(const float* __restrict__ ipre,  // [B][3*hid]
                              const float* __restrict__ hpre,  // [B][3*hid]
                              const float* __restrict__ lig, const float* __restrict__ lib,
                              const float* __restrict__ lhg, const float* __restrict__ lhb,
                              float* __restrict__ h,           // [B][hid]
                              __hip_bfloat16* __restrict__ hbf,
                              int hid) {
  __shared__ float red[256];
  __shared__ float st[12];   // st[0..5] = means (i0,i1,i2,h0,h1,h2); st[6..11] = rstds
  const int row = blockIdx.x;
  const float* ip = ipre + (size_t)row * 3 * hid;
  const float* hp = hpre + (size_t)row * 3 * hid;
  float* hr = h + (size_t)row * hid;

  for (int q = 0; q < 6; ++q) {
    const float* p = (q < 3) ? (ip + q * hid) : (hp + (q - 3) * hid);
    float sum = 0.f, sq = 0.f;
    for (int c = threadIdx.x; c < hid; c += 256) {
      float v = p[c];
      sum += v;
      sq  += v * v;
    }
    sum = blockReduceSum(sum, red);
    sq  = blockReduceSum(sq, red);
    if (threadIdx.x == 0) {
      float m = sum / (float)hid;
      st[q]     = m;
      st[q + 6] = rsqrtf(sq / (float)hid - m * m + EPS_);
    }
    __syncthreads();
  }

  for (int c = threadIdx.x; c < hid; c += 256) {
    float iv[3], hv[3];
#pragma unroll
    for (int g = 0; g < 3; ++g) {
      iv[g] = (ip[g * hid + c] - st[g])     * st[g + 6] * lig[g * hid + c] + lib[g * hid + c];
      hv[g] = (hp[g * hid + c] - st[3 + g]) * st[9 + g] * lhg[g * hid + c] + lhb[g * hid + c];
    }
    const float r = 1.f / (1.f + __expf(-(iv[0] + hv[0])));
    const float z = 1.f / (1.f + __expf(-(iv[1] + hv[1])));
    const float n = tanhf(iv[2] + r * hv[2]);
    const float nh = (1.f - z) * n + z * hr[c];
    hr[c] = nh;
    hbf[(size_t)row * hid + c] = __float2bfloat16(nh);
  }
}

// ---------------------------------------------------------------------------
extern "C" void kernel_launch(void* const* d_in, const int* in_sizes, int n_in,
                              void* d_out, int out_size, void* d_ws, size_t ws_size,
                              hipStream_t stream) {
  (void)in_sizes; (void)n_in; (void)out_size; (void)ws_size;

  const int*   ids   = (const int*)  d_in[0];
  const float* emb   = (const float*)d_in[1];
  const float* lng   = (const float*)d_in[2];
  const float* lnb   = (const float*)d_in[3];
  const float* lowWi = (const float*)d_in[4];
  const float* lowbi = (const float*)d_in[5];
  const float* lowWh = (const float*)d_in[6];
  const float* lowbh = (const float*)d_in[7];
  const float* l_ig  = (const float*)d_in[8];
  const float* l_ib  = (const float*)d_in[9];
  const float* l_hg  = (const float*)d_in[10];
  const float* l_hb  = (const float*)d_in[11];
  const float* hiWi  = (const float*)d_in[12];
  const float* hibi  = (const float*)d_in[13];
  const float* hiWh  = (const float*)d_in[14];
  const float* hibh  = (const float*)d_in[15];
  const float* h_ig  = (const float*)d_in[16];
  const float* h_ib  = (const float*)d_in[17];
  const float* h_hg  = (const float*)d_in[18];
  const float* h_hb  = (const float*)d_in[19];
  const float* Wout  = (const float*)d_in[20];
  const float* bout  = (const float*)d_in[21];
  float* out = (float*)d_out;

  char* ws = (char*)d_ws;
  size_t off = 0;
  auto alloc = [&](size_t bytes) -> void* {
    void* p = ws + off;
    off += (bytes + 255) & ~(size_t)255;
    return p;
  };

  __hip_bfloat16* lowWiT = (__hip_bfloat16*)alloc((size_t)3 * H_  * LOWIN_ * 2);
  __hip_bfloat16* lowWhT = (__hip_bfloat16*)alloc((size_t)3 * H_  * H_     * 2);
  __hip_bfloat16* hiWiT  = (__hip_bfloat16*)alloc((size_t)3 * HH_ * H_     * 2);
  __hip_bfloat16* hiWhT  = (__hip_bfloat16*)alloc((size_t)3 * HH_ * HH_    * 2);
  __hip_bfloat16* WoutT  = (__hip_bfloat16*)alloc((size_t)V_ * HH_ * 2);
  __hip_bfloat16* xbf    = (__hip_bfloat16*)alloc((size_t)S_ * B_ * E_ * 2);
  __hip_bfloat16* xcat   = (__hip_bfloat16*)alloc((size_t)B_ * LOWIN_ * 2);
  float*          hl     = (float*)         alloc((size_t)B_ * H_  * 4);
  __hip_bfloat16* hlb    = (__hip_bfloat16*)alloc((size_t)B_ * H_  * 2);
  float*          hh     = (float*)         alloc((size_t)B_ * HH_ * 4);
  __hip_bfloat16* hhb    = (__hip_bfloat16*)alloc((size_t)B_ * HH_ * 2);
  float*          ipre   = (float*)         alloc((size_t)B_ * 3 * H_ * 4);
  float*          hpre   = (float*)         alloc((size_t)B_ * 3 * H_ * 4);

  const dim3 tb(32, 8);
  // Weight convert + transpose (once per launch; stays hot in 192MB L2 after)
  k_transpose_bf16<<<dim3((3 * H_) / 32, LOWIN_ / 32), tb, 0, stream>>>(lowWi, lowWiT, LOWIN_, 3 * H_);
  k_transpose_bf16<<<dim3((3 * H_) / 32, H_ / 32),     tb, 0, stream>>>(lowWh, lowWhT, H_,     3 * H_);
  k_transpose_bf16<<<dim3((3 * HH_) / 32, H_ / 32),    tb, 0, stream>>>(hiWi,  hiWiT,  H_,     3 * HH_);
  k_transpose_bf16<<<dim3((3 * HH_) / 32, HH_ / 32),   tb, 0, stream>>>(hiWh,  hiWhT,  HH_,    3 * HH_);
  k_transpose_bf16<<<dim3(V_ / 32, HH_ / 32),          tb, 0, stream>>>(Wout,  WoutT,  HH_,    V_);

  k_embed_ln<<<B_ * S_, 256, 0, stream>>>(ids, emb, lng, lnb, xbf);
  k_zero_state<<<(B_ * H_ + 255) / 256, 256, 0, stream>>>(hl, hlb, hh, hhb);

  for (int s = 0; s < S_; ++s) {
    const __hip_bfloat16* xs = xbf + (size_t)s * B_ * E_;
    for (int n = 0; n < 3; ++n) {
      k_pack_xcat<<<B_, 256, 0, stream>>>(xs, hhb, xcat);
      for (int t = 0; t < 5; ++t) {
        k_gemm_bf16_wmma<<<(3 * H_) / 64, 128, 0, stream>>>(xcat, lowWiT, lowbi, ipre, LOWIN_, 3 * H_);
        k_gemm_bf16_wmma<<<(3 * H_) / 64, 128, 0, stream>>>(hlb,  lowWhT, lowbh, hpre, H_,     3 * H_);
        k_cell_update<<<B_, 256, 0, stream>>>(ipre, hpre, l_ig, l_ib, l_hg, l_hb, hl, hlb, H_);
      }
      k_gemm_bf16_wmma<<<(3 * HH_) / 64, 128, 0, stream>>>(hlb, hiWiT, hibi, ipre, H_,  3 * HH_);
      k_gemm_bf16_wmma<<<(3 * HH_) / 64, 128, 0, stream>>>(hhb, hiWhT, hibh, hpre, HH_, 3 * HH_);
      k_cell_update<<<B_, 256, 0, stream>>>(ipre, hpre, h_ig, h_ib, h_hg, h_hb, hh, hhb, HH_);
    }
    // logits for step s written straight into d_out[b][s][v] (row stride S*V)
    k_gemm_bf16_wmma<<<V_ / 64, 128, 0, stream>>>(hhb, WoutT, bout, out + (size_t)s * V_, HH_, (long)S_ * V_);
  }
}
